// AdaptiveIncidenceAttention_50989851738124
// MI455X (gfx1250) — matrix operand order repaired and checked by reading, not compiled, not run on hardware
//
#include <hip/hip_runtime.h>
#include <hip/hip_bf16.h>
#include <cstddef>

// ---------------------------------------------------------------------------
// AdaptiveIncidenceAttention for MI455X (gfx1250, wave32, WMMA)
//   1) context mean + tiny MLP -> per-batch scale
//   2) cast x -> f16; transpose+cast Wqkv, Wo -> f16
//   3) QKV GEMM (v_wmma_f32_16x16x32_f16, 32x64 tile/wave), bias fused
//   4) transpose V -> vT[B,H,dh,S] f16
//   5) flash attention: 8 waves/block share K/V via double-buffered LDS;
//      16 q-rows/wave; rowsum via WMMA-with-ones; swizzle max; rcp/exp2
//   6) output GEMM with bias + adaptive scale, f32 out
// ---------------------------------------------------------------------------

typedef __attribute__((ext_vector_type(16))) _Float16 v16h;
typedef __attribute__((ext_vector_type(8)))  _Float16 v8h;
typedef __attribute__((ext_vector_type(8)))  float    v8f;

union AU { v16h v; v8h h[2]; };

__device__ __forceinline__ v8f wmma16(v16h a, v16h b, v8f c) {
  return __builtin_amdgcn_wmma_f32_16x16x32_f16(false, a, false, b,
                                                (short)0, c, false, false);
}

#define ZERO8 ((v8f){0.f,0.f,0.f,0.f,0.f,0.f,0.f,0.f})
#define L2E   1.44269504088896340736f

__device__ __forceinline__ float fastexp(float x) { return __builtin_amdgcn_exp2f(x * L2E); }
__device__ __forceinline__ float fastrcp(float x) { return __builtin_amdgcn_rcpf(x); }

// xor-swizzle max across 16-lane groups (imm: and=0x1F, or=0, xor<<10)
template <int IMM>
__device__ __forceinline__ float swz_max(float v) {
  return fmaxf(v, __int_as_float(__builtin_amdgcn_ds_swizzle(__float_as_int(v), IMM)));
}

// Problem constants
#define BB   2
#define SS   2048
#define DD   1024
#define HH   16
#define DH   64
#define MROWS (BB*SS)        // 4096
#define QKVN (3*DD)          // 3072

// --------------------------- small prologue kernels ------------------------

__global__ void context_mean_kernel(const float* __restrict__ x, float* __restrict__ ctx) {
  int i = blockIdx.x * 256 + threadIdx.x;          // B*D = 2048 threads
  if (i >= BB * DD) return;
  int d = i & (DD - 1);
  int b = i >> 10;
  const float* p = x + (size_t)b * SS * DD + d;
  float s = 0.f;
  for (int t = 0; t < SS; ++t) s += p[(size_t)t * DD];
  ctx[i] = s * (1.0f / (float)SS);
}

__global__ void mlp_scale_kernel(const float* __restrict__ ctx,
                                 const float* __restrict__ W1, const float* __restrict__ b1,
                                 const float* __restrict__ W2, const float* __restrict__ b2,
                                 const float* __restrict__ aw,
                                 float* __restrict__ scale) {
  __shared__ float sh_ctx[DD];
  __shared__ float sh_h[DD / 2];
  __shared__ float sh_f[HH];
  const int b = blockIdx.x;
  const int t = threadIdx.x;                       // 512 threads
  sh_ctx[t]       = ctx[b * DD + t];
  sh_ctx[t + 512] = ctx[b * DD + t + 512];
  __syncthreads();
  float acc = b1[t];
  for (int k = 0; k < DD; ++k) acc += sh_ctx[k] * W1[(size_t)k * (DD / 2) + t];
  sh_h[t] = 0.5f * acc * (1.0f + erff(acc * 0.70710678118654752f));
  __syncthreads();
  if (t < HH) {
    float a2 = b2[t];
    for (int k = 0; k < DD / 2; ++k) a2 += sh_h[k] * W2[(size_t)k * HH + t];
    sh_f[t] = fastrcp(1.0f + fastexp(-a2));
  }
  __syncthreads();
  if (t == 0) {
    float s = 0.f;
    for (int k = 0; k < HH; ++k) s += sh_f[k];
    scale[b] = 1.0f + aw[0] * (s * (1.0f / (float)HH));
  }
}

__global__ void cast_f16_kernel(const float* __restrict__ in, _Float16* __restrict__ out, int n) {
  int i = blockIdx.x * 256 + threadIdx.x;
  if (i < n) out[i] = (_Float16)in[i];
}

// Wt[n][k] = (f16) W[k][n]   (W is [K,N] row-major)
__global__ void transpose_cast_kernel(const float* __restrict__ W, _Float16* __restrict__ Wt,
                                      int K, int N) {
  int i = blockIdx.x * 256 + threadIdx.x;
  if (i >= N * K) return;
  int n = i / K, k = i - n * K;
  Wt[i] = (_Float16)W[(size_t)k * N + n];
}

// vT[((b*H+h)*DH + d)*S + s] = qkv[(b*S+s)*3072 + 2048 + h*64 + d]
__global__ void transpose_v_kernel(const _Float16* __restrict__ qkv, _Float16* __restrict__ vT) {
  int i = blockIdx.x * 256 + threadIdx.x;          // B*H*DH*S = 4,194,304
  if (i >= BB * HH * DH * SS) return;
  int s = i & (SS - 1);
  int d = (i >> 11) & (DH - 1);
  int h = (i >> 17) & (HH - 1);
  int b = i >> 21;
  vT[i] = qkv[(size_t)(b * SS + s) * QKVN + 2 * DD + h * DH + d];
}

// ------------------------------- WMMA GEMM ---------------------------------
// C[M,N] = A[M,K](f16) @ Bt[N,K]^T(f16) + bias[N]
// Wave computes a 32x64 tile (2 M-tiles x 4 N-tiles -> 8 wmma / K-chunk).
// Block = 4 waves arranged 2x2 -> 64(M) x 128(N). grid: (N/128, M/64).
template <bool OUT_F16>
__global__ __launch_bounds__(128)
void gemm_kernel(const _Float16* __restrict__ A,
                 const _Float16* __restrict__ Bt,
                 const float* __restrict__ bias,
                 void* __restrict__ Cout,
                 int N, int K,
                 const float* __restrict__ scale, int rowsPerBatch) {
  const int lane  = threadIdx.x & 31;
  const int wave  = threadIdx.x >> 5;
  const int lhalf = lane >> 4;
  const int l16   = lane & 15;
  const int tileM = blockIdx.y * 64  + (wave >> 1) * 32;
  const int tileN = blockIdx.x * 128 + (wave & 1) * 64;

  const _Float16* arow[2];
  arow[0] = A + (size_t)(tileM + l16) * K + lhalf * 16;
  arow[1] = arow[0] + (size_t)16 * K;
  const _Float16* brow[4];
#pragma unroll
  for (int ni = 0; ni < 4; ++ni)
    brow[ni] = Bt + (size_t)(tileN + ni * 16 + l16) * K + lhalf * 16;

  v8f acc[2][4];
#pragma unroll
  for (int mi = 0; mi < 2; ++mi)
#pragma unroll
    for (int ni = 0; ni < 4; ++ni) acc[mi][ni] = ZERO8;

  for (int k0 = 0; k0 < K; k0 += 32) {
    __builtin_prefetch(arow[0] + k0 + 512, 0, 0);
    AU a[2];
#pragma unroll
    for (int mi = 0; mi < 2; ++mi) {
      a[mi].h[0] = *(const v8h*)(arow[mi] + k0);
      a[mi].h[1] = *(const v8h*)(arow[mi] + k0 + 8);
    }
    v16h bv[4];
#pragma unroll
    for (int ni = 0; ni < 4; ++ni) bv[ni] = *(const v16h*)(brow[ni] + k0);
#pragma unroll
    for (int mi = 0; mi < 2; ++mi)
#pragma unroll
      for (int ni = 0; ni < 4; ++ni)
        acc[mi][ni] = wmma16(a[mi].v, bv[ni], acc[mi][ni]);
  }

#pragma unroll
  for (int ni = 0; ni < 4; ++ni) {
    const float bcol = bias[tileN + ni * 16 + l16];
#pragma unroll
    for (int mi = 0; mi < 2; ++mi) {
      if (OUT_F16) {
        _Float16* C = (_Float16*)Cout;
#pragma unroll
        for (int r = 0; r < 8; ++r) {
          int row = tileM + mi * 16 + r + lhalf * 8;
          C[(size_t)row * N + tileN + ni * 16 + l16] = (_Float16)(acc[mi][ni][r] + bcol);
        }
      } else {
        float* C = (float*)Cout;
#pragma unroll
        for (int r = 0; r < 8; ++r) {
          int row = tileM + mi * 16 + r + lhalf * 8;
          float sc = scale[row / rowsPerBatch];
          C[(size_t)row * N + tileN + ni * 16 + l16] = (acc[mi][ni][r] + bcol) * sc;
        }
      }
    }
  }
}

// ------------------------------ attention ----------------------------------
// Block = 8 waves (256 threads), all on one (b,h), q-rows [qblk*128, +128).
// Each wave owns a 16-row q tile. K (32x64) and V (64x32) tiles are staged
// into double-buffered LDS cooperatively each 32-k iteration.
#define KLDSW 72   // K tile row stride in f16 (144B: 16B-mult, conflict-free)
#define VLDSW 40   // V tile row stride in f16 (80B)
#define PLDSW 40   // P tile row stride in f16

__global__ __launch_bounds__(256)
void attention_kernel(const _Float16* __restrict__ qkv,  // [B*S, 3072]
                      const _Float16* __restrict__ vT,   // [B*H*DH, S]
                      const float* __restrict__ mobius,  // [H]
                      _Float16* __restrict__ out) {      // [B*S, D]
  __shared__ _Float16 ldsK[2][32 * KLDSW];   //  9216 B
  __shared__ _Float16 ldsV[2][64 * VLDSW];   // 10240 B
  __shared__ _Float16 ldsP[8][16 * PLDSW];   // 10240 B
  const int tid   = threadIdx.x;
  const int wave  = tid >> 5;
  const int lane  = tid & 31;
  const int lhalf = lane >> 4;
  const int l16   = lane & 15;

  const int qblk = blockIdx.x & 15;                // S/128 = 16
  const int h    = (blockIdx.x >> 4) & 15;
  const int b    = blockIdx.x >> 8;
  const int qbase = qblk * 128 + wave * 16;

  const _Float16* qb  = qkv + (size_t)(b * SS) * QKVN + h * DH;
  const _Float16* kb_ = qb + DD;
  const _Float16* vTb = vT + (size_t)((b * HH + h) * DH) * SS;

  // Q A-operands (2 dh-chunks), per wave
  AU qa[2];
  {
    const _Float16* qrow = qb + (size_t)(qbase + l16) * QKVN;
#pragma unroll
    for (int c = 0; c < 2; ++c) {
      const _Float16* p = qrow + c * 32 + lhalf * 16;
      qa[c].h[0] = *(const v8h*)(p);
      qa[c].h[1] = *(const v8h*)(p + 8);
    }
  }

  v16h ones;
#pragma unroll
  for (int i = 0; i < 16; ++i) ones[i] = (_Float16)1.0f;

  float m[8];
  v8f lsum = ZERO8;
  v8f o[4];
#pragma unroll
  for (int r = 0; r < 8; ++r) m[r] = -1e30f;
#pragma unroll
  for (int t = 0; t < 4; ++t) o[t] = ZERO8;
  const float ms = mobius[h];

  // cooperative staging addresses (per thread)
  const int krow = tid >> 3, kseg = tid & 7;       // 32 rows x 8 segs of 8 f16
  const int vrow = tid >> 2, vseg = tid & 3;       // 64 rows x 4 segs of 8 f16

  for (int kb = 0; kb < SS; kb += 32) {
    const int buf = (kb >> 5) & 1;
    // ---- stage K tile (32 x 64 f16) and V tile (64 x 32 f16) into LDS
    *(v8h*)&ldsK[buf][krow * KLDSW + kseg * 8] =
        *(const v8h*)(kb_ + (size_t)(kb + krow) * QKVN + kseg * 8);
    *(v8h*)&ldsV[buf][vrow * VLDSW + vseg * 8] =
        *(const v8h*)(vTb + (size_t)vrow * SS + kb + vseg * 8);
    __syncthreads();

    // ---- scores: two 16x16 tiles from LDS K
    v8f s[2];
#pragma unroll
    for (int t = 0; t < 2; ++t) {
      v8f c = ZERO8;
#pragma unroll
      for (int cch = 0; cch < 2; ++cch) {
        v16h kop = *(const v16h*)&ldsK[buf][(t * 16 + l16) * KLDSW + cch * 32 + lhalf * 16];
        c = wmma16(qa[cch].v, kop, c);
      }
      s[t] = c;
    }

    // ---- mobius + online softmax (swizzle max, rcp/exp2)
#pragma unroll
    for (int r = 0; r < 8; ++r) {
      float s0 = s[0][r] * 0.125f;                 // 1/sqrt(64)
      s0 += ms * s0 * fastrcp(1.0f + s0 * s0);
      float s1 = s[1][r] * 0.125f;
      s1 += ms * s1 * fastrcp(1.0f + s1 * s1);

      float tmax = fmaxf(s0, s1);
      tmax = swz_max<0x041F>(tmax);                // xor 1
      tmax = swz_max<0x081F>(tmax);                // xor 2
      tmax = swz_max<0x101F>(tmax);                // xor 4
      tmax = swz_max<0x201F>(tmax);                // xor 8

      float mn    = fmaxf(m[r], tmax);
      float alpha = fastexp(m[r] - mn);
      m[r] = mn;
      float p0 = fastexp(s0 - mn);
      float p1 = fastexp(s1 - mn);

      lsum[r] *= alpha;
#pragma unroll
      for (int t4 = 0; t4 < 4; ++t4) o[t4][r] *= alpha;

      const int prow = (r + lhalf * 8) * PLDSW;
      ldsP[wave][prow + l16]      = (_Float16)p0;
      ldsP[wave][prow + 16 + l16] = (_Float16)p1;
    }
    asm volatile("s_wait_dscnt 0x0" ::: "memory");  // wave-local P RAW

    // ---- P A-operand; rowsum via WMMA with ones; P@V from LDS V
    AU pa;
    {
      const _Float16* p = &ldsP[wave][l16 * PLDSW + lhalf * 16];
      pa.h[0] = *(const v8h*)(p);
      pa.h[1] = *(const v8h*)(p + 8);
    }
    lsum = wmma16(pa.v, ones, lsum);
#pragma unroll
    for (int t = 0; t < 4; ++t) {
      v16h vb = *(const v16h*)&ldsV[buf][(t * 16 + l16) * VLDSW + lhalf * 16];
      o[t] = wmma16(pa.v, vb, o[t]);
    }
    // no trailing barrier: next iteration writes the other LDS buffer
  }

  // ---- epilogue: normalize (rcp) and store f16 [B*S, D]
#pragma unroll
  for (int r = 0; r < 8; ++r) {
    float rinv = fastrcp(lsum[r]);
    int row = b * SS + qbase + r + lhalf * 8;
#pragma unroll
    for (int t = 0; t < 4; ++t)
      out[(size_t)row * DD + h * DH + t * 16 + l16] = (_Float16)(o[t][r] * rinv);
  }
}

// ------------------------------- launcher ----------------------------------

extern "C" void kernel_launch(void* const* d_in, const int* in_sizes, int n_in,
                              void* d_out, int out_size, void* d_ws, size_t ws_size,
                              hipStream_t stream) {
  (void)in_sizes; (void)n_in; (void)out_size; (void)ws_size;

  const float* x      = (const float*)d_in[0];
  const float* Wqkv   = (const float*)d_in[1];
  const float* bqkv   = (const float*)d_in[2];
  const float* Wo     = (const float*)d_in[3];
  const float* bo     = (const float*)d_in[4];
  const float* mobius = (const float*)d_in[5];
  const float* W1     = (const float*)d_in[6];
  const float* b1     = (const float*)d_in[7];
  const float* W2     = (const float*)d_in[8];
  const float* b2     = (const float*)d_in[9];
  const float* aw     = (const float*)d_in[10];
  float* outp         = (float*)d_out;

  char* w = (char*)d_ws;
  auto carve = [&](size_t bytes) {
    void* p = (void*)w;
    w += (bytes + 255) & ~(size_t)255;
    return p;
  };
  float*     ctx   = (float*)    carve((size_t)BB * DD * sizeof(float));
  float*     scl   = (float*)    carve((size_t)BB * sizeof(float));
  _Float16*  xh    = (_Float16*) carve((size_t)MROWS * DD * 2);
  _Float16*  wqkvT = (_Float16*) carve((size_t)QKVN * DD * 2);
  _Float16*  woT   = (_Float16*) carve((size_t)DD * DD * 2);
  _Float16*  qkvh  = (_Float16*) carve((size_t)MROWS * QKVN * 2);
  _Float16*  vTh   = (_Float16*) carve((size_t)BB * HH * DH * SS * 2);
  _Float16*  attnh = (_Float16*) carve((size_t)MROWS * DD * 2);

  context_mean_kernel<<<(BB * DD + 255) / 256, 256, 0, stream>>>(x, ctx);
  mlp_scale_kernel<<<BB, 512, 0, stream>>>(ctx, W1, b1, W2, b2, aw, scl);

  {
    int n = MROWS * DD;
    cast_f16_kernel<<<(n + 255) / 256, 256, 0, stream>>>(x, xh, n);
  }
  transpose_cast_kernel<<<(QKVN * DD + 255) / 256, 256, 0, stream>>>(Wqkv, wqkvT, DD, QKVN);
  transpose_cast_kernel<<<(DD * DD + 255) / 256, 256, 0, stream>>>(Wo, woT, DD, DD);

  {
    dim3 grid(QKVN / 128, MROWS / 64);
    gemm_kernel<true><<<grid, 128, 0, stream>>>(xh, wqkvT, bqkv, (void*)qkvh,
                                                QKVN, DD, scl, SS);
  }

  {
    int n = BB * HH * DH * SS;
    transpose_v_kernel<<<(n + 255) / 256, 256, 0, stream>>>(qkvh, vTh);
  }

  // B*H*(S/128) = 512 blocks, 256 threads
  attention_kernel<<<BB * HH * (SS / 128), 256, 0, stream>>>(qkvh, vTh, mobius, attnh);

  {
    dim3 grid(DD / 128, MROWS / 64);
    gemm_kernel<false><<<grid, 128, 0, stream>>>(attnh, woT, bo, (void*)outp,
                                                 DD, DD, scl, SS);
  }
}
